// GatedMultiHeadAttention_61014305407004
// MI455X (gfx1250) — matrix operand order repaired and checked by reading, not compiled, not run on hardware
//
#include <hip/hip_runtime.h>

typedef __bf16 bf16;
typedef __attribute__((ext_vector_type(16))) __bf16 v16bf;
typedef __attribute__((ext_vector_type(8)))  float  v8f;

#define D_MODEL 768
#define NHEAD   12
#define D_HEAD  64
#define SEQ     4096
#define BATCH   2
#define NTOK    (BATCH * SEQ)

#define AS_STRIDE 40   // padded LDS row stride (bf16 elems) for 32-wide k tiles
#define KS_STRIDE 72   // padded LDS row stride (bf16 elems) for 64-wide rows

// A-fragment (16x32, 16-bit) k-index for element e (0..15) given lane-half.
__device__ __forceinline__ int kmapA(int e, int lhalf) {
    return lhalf * 16 + e;
}

// LDS byte address (flat aperture keeps the LDS offset in addr[31:0]).
__device__ __forceinline__ unsigned ldsa(const void* p) {
    return (unsigned)(size_t)p;
}

// Async global->LDS copy of 32 bytes per lane (two b128 ops; the instruction
// offset is applied to BOTH the LDS and the memory address).
__device__ __forceinline__ void async_cp32(unsigned lds, const void* gaddr) {
    asm volatile("global_load_async_to_lds_b128 %0, %1, off"
                 :: "v"(lds), "v"(gaddr) : "memory");
    asm volatile("global_load_async_to_lds_b128 %0, %1, off offset:16"
                 :: "v"(lds), "v"(gaddr) : "memory");
}

// ---------------------------------------------------------------------------
// fp32 -> bf16 conversion (grid-stride)
// ---------------------------------------------------------------------------
__global__ void cvt_f32_bf16(const float* __restrict__ src, bf16* __restrict__ dst, int n) {
    int i = blockIdx.x * blockDim.x + threadIdx.x;
    int stride = gridDim.x * blockDim.x;
    for (; i < n; i += stride) dst[i] = (bf16)src[i];
}

// ---------------------------------------------------------------------------
// fp32 [K x N] -> bf16 transposed [N x K]; grid (N/32, K/32), 256 threads.
// ---------------------------------------------------------------------------
__global__ __launch_bounds__(256) void cvt_transpose(
    const float* __restrict__ src, bf16* __restrict__ dst, int K, int N) {
    __shared__ float tile[32][33];
    const int bx = blockIdx.x * 32;  // n
    const int by = blockIdx.y * 32;  // k
    const int tx = threadIdx.x & 31;
    const int ty = threadIdx.x >> 5;  // 0..7
    for (int i = 0; i < 32; i += 8)
        tile[ty + i][tx] = src[(size_t)(by + ty + i) * N + bx + tx];
    __syncthreads();
    for (int i = 0; i < 32; i += 8)
        dst[(size_t)(bx + ty + i) * K + by + tx] = (bf16)tile[tx][ty + i];
}

// ---------------------------------------------------------------------------
// bf16 WMMA GEMM: C[MxN] = A[MxK] * Bt[NxK]^T  (Bt is N-major / pre-transposed).
// Block = 256 thr (8 waves), tile 128x128, K-step 32, double-buffered LDS
// filled with async global->LDS DMA (ASYNCcnt pipelining).
// EPI: 0 = bf16 store to C0[M][N]
//      1 = split: n<768 -> bf16 q to C0, else f32 gate to C1 (row stride 768)
//      2 = f32 store to C0[M][N]
//      3 = bf16 transposed store to C0[N][M]
// ---------------------------------------------------------------------------
template <int EPI>
__global__ __launch_bounds__(256) void gemm_bf16(
    const bf16* __restrict__ A, const bf16* __restrict__ Bt,
    void* __restrict__ C0, void* __restrict__ C1, int M, int N, int K) {
    __shared__ alignas(16) bf16 As[2][128 * AS_STRIDE];
    __shared__ alignas(16) bf16 Bs[2][128 * AS_STRIDE];

    const int tid   = threadIdx.x;
    const int lane  = tid & 31;
    const int wave  = tid >> 5;
    const int lhalf = (lane >> 4) & 1;
    const int l16   = lane & 15;
    const int wm    = wave >> 1;  // 0..3  (M direction, 32 rows each)
    const int wn    = wave & 1;   // 0..1  (N direction, 64 cols each)
    const int m0    = blockIdx.y * 128;
    const int n0    = blockIdx.x * 128;

    v8f acc[2][4] = {};

    const int trow = tid >> 1, tcol = (tid & 1) * 16;  // 128x32 tile, 16 elem/thr
    const bf16* gA = A  + (size_t)(m0 + trow) * K + tcol;
    const bf16* gB = Bt + (size_t)(n0 + trow) * K + tcol;
    const unsigned lA[2] = { ldsa(&As[0][trow * AS_STRIDE + tcol]),
                             ldsa(&As[1][trow * AS_STRIDE + tcol]) };
    const unsigned lB[2] = { ldsa(&Bs[0][trow * AS_STRIDE + tcol]),
                             ldsa(&Bs[1][trow * AS_STRIDE + tcol]) };

    auto issue = [&](int k0, int buf) {
        async_cp32(lA[buf], gA + k0);
        async_cp32(lB[buf], gB + k0);
    };

    issue(0, 0);
    const int nk = K / 32;
    for (int i = 0; i < nk; ++i) {
        if (i + 1 < nk) {
            issue((i + 1) * 32, (i + 1) & 1);
            asm volatile("s_wait_asynccnt 0x4" ::: "memory");  // tile i complete
        } else {
            asm volatile("s_wait_asynccnt 0x0" ::: "memory");
        }
        __syncthreads();

        const bf16* as = As[i & 1];
        const bf16* bs = Bs[i & 1];
        v16bf afr[2];
        for (int t = 0; t < 2; ++t) {
            const bf16* ap = as + (wm * 32 + t * 16 + l16) * AS_STRIDE;
            for (int e = 0; e < 16; ++e) afr[t][e] = ap[kmapA(e, lhalf)];
        }
        v16bf bfr[4];
        for (int u = 0; u < 4; ++u) {
            const bf16* bp = bs + (wn * 64 + u * 16 + l16) * AS_STRIDE + lhalf * 16;
            for (int e = 0; e < 16; ++e) bfr[u][e] = bp[e];
        }
        for (int t = 0; t < 2; ++t)
            for (int u = 0; u < 4; ++u)
                acc[t][u] = __builtin_amdgcn_wmma_f32_16x16x32_bf16(
                    false, afr[t], false, bfr[u], (short)0, acc[t][u], false, false);
        __syncthreads();  // all waves done with buf[i&1] before tile i+2 lands
    }

    for (int t = 0; t < 2; ++t) {
        const int mbase = m0 + wm * 32 + t * 16 + lhalf * 8;
        for (int u = 0; u < 4; ++u) {
            const int n = n0 + wn * 64 + u * 16 + l16;
            if (EPI == 3) {
                bf16 tmp[8];
                for (int r = 0; r < 8; ++r) tmp[r] = (bf16)acc[t][u][r];
                *(uint4*)((bf16*)C0 + (size_t)n * M + mbase) = *(const uint4*)tmp;
            } else {
                for (int r = 0; r < 8; ++r) {
                    const float val = acc[t][u][r];
                    const size_t mm = (size_t)(mbase + r);
                    if (EPI == 0) {
                        ((bf16*)C0)[mm * N + n] = (bf16)val;
                    } else if (EPI == 1) {
                        if (n < D_MODEL) ((bf16*)C0)[mm * D_MODEL + n] = (bf16)val;
                        else             ((float*)C1)[mm * D_MODEL + (n - D_MODEL)] = val;
                    } else {
                        ((float*)C0)[mm * N + n] = val;
                    }
                }
            }
        }
    }
}

// ---------------------------------------------------------------------------
// Flash attention + gate. Grid: (SEQ/128, BATCH*NHEAD), 256 threads.
// Wave w owns q rows [qblk*128 + w*16, +16). K/V chunks of 64 keys staged in
// double-buffered LDS via async global->LDS DMA.
// vT is the pre-transposed value matrix [D_MODEL][NTOK].
// ---------------------------------------------------------------------------
__global__ __launch_bounds__(256) void attn_kernel(
    const bf16* __restrict__ qh, const bf16* __restrict__ kh,
    const bf16* __restrict__ vT, const float* __restrict__ gate,
    bf16* __restrict__ oh) {
    __shared__ alignas(16) bf16 Ks[2][64 * KS_STRIDE];     // [key][dim]
    __shared__ alignas(16) bf16 Vt[2][64 * KS_STRIDE];     // [dim][key]
    __shared__ alignas(16) bf16 Ps[8][16 * KS_STRIDE];     // per-wave P bounce

    const int tid   = threadIdx.x;
    const int lane  = tid & 31;
    const int wave  = tid >> 5;
    const int lhalf = (lane >> 4) & 1;
    const int l16   = lane & 15;

    const int b = blockIdx.y / NHEAD;
    const int h = blockIdx.y % NHEAD;
    const size_t tokBase = (size_t)b * SEQ;
    const int qrow0 = blockIdx.x * 128 + wave * 16;

    // Q fragments, resident for whole kernel (M=16, K=64 -> 2 chunks of 32)
    v16bf qf[2];
    {
        const bf16* qp = qh + (tokBase + qrow0 + l16) * D_MODEL + h * D_HEAD;
        for (int kc = 0; kc < 2; ++kc)
            for (int e = 0; e < 16; ++e)
                qf[kc][e] = qp[kc * 32 + kmapA(e, lhalf)];
    }

    v8f o[4] = {};
    float mrow[8], lrow[8];
    for (int r = 0; r < 8; ++r) { mrow[r] = -1e30f; lrow[r] = 0.f; }

    const int srow = tid >> 2, scol = (tid & 3) * 16;  // 64x64 stage, 16 elem/thr
    const float scale = 0.125f;                        // 1/sqrt(64)
    bf16* ps = Ps[wave];

    const bf16* gK = kh + (tokBase + srow) * D_MODEL + h * D_HEAD + scol;
    const bf16* gV = vT + (size_t)(h * D_HEAD + srow) * NTOK + tokBase + scol;
    const unsigned lK[2] = { ldsa(&Ks[0][srow * KS_STRIDE + scol]),
                             ldsa(&Ks[1][srow * KS_STRIDE + scol]) };
    const unsigned lV[2] = { ldsa(&Vt[0][srow * KS_STRIDE + scol]),
                             ldsa(&Vt[1][srow * KS_STRIDE + scol]) };

    auto issue = [&](int j0, int buf) {
        async_cp32(lK[buf], gK + (size_t)j0 * D_MODEL);  // 64 keys x 64 dims
        async_cp32(lV[buf], gV + j0);                    // 64 dims x 64 keys
    };

    issue(0, 0);
    const int nchunk = SEQ / 64;
    for (int i = 0; i < nchunk; ++i) {
        if (i + 1 < nchunk) {
            issue((i + 1) * 64, (i + 1) & 1);
            asm volatile("s_wait_asynccnt 0x4" ::: "memory");  // chunk i complete
        } else {
            asm volatile("s_wait_asynccnt 0x0" ::: "memory");
        }
        __syncthreads();
        const bf16* ks = Ks[i & 1];
        const bf16* vt = Vt[i & 1];

        // scores = Q @ K^T  (M=16, N=64 keys, K=64 dims)
        v8f s[4] = {};
        for (int kc = 0; kc < 2; ++kc) {
            for (int u = 0; u < 4; ++u) {
                v16bf bfr;
                const bf16* bp = ks + (u * 16 + l16) * KS_STRIDE + kc * 32 + lhalf * 16;
                for (int e = 0; e < 16; ++e) bfr[e] = bp[e];
                s[u] = __builtin_amdgcn_wmma_f32_16x16x32_bf16(
                    false, qf[kc], false, bfr, (short)0, s[u], false, false);
            }
        }

        // online softmax (row m = r + 8*lhalf lives in element r, cols across 16 lanes)
        for (int u = 0; u < 4; ++u)
            for (int r = 0; r < 8; ++r) s[u][r] *= scale;

        float alpha[8];
        for (int r = 0; r < 8; ++r) {
            float rm = fmaxf(fmaxf(s[0][r], s[1][r]), fmaxf(s[2][r], s[3][r]));
            for (int mk = 1; mk < 16; mk <<= 1) rm = fmaxf(rm, __shfl_xor(rm, mk, 32));
            const float mnew = fmaxf(mrow[r], rm);
            alpha[r] = __expf(mrow[r] - mnew);
            mrow[r] = mnew;
        }
        float rsum[8] = {};
        for (int u = 0; u < 4; ++u)
            for (int r = 0; r < 8; ++r) {
                const float p = __expf(s[u][r] - mrow[r]);
                s[u][r] = p;
                rsum[r] += p;
            }
        for (int r = 0; r < 8; ++r) {
            float rs = rsum[r];
            for (int mk = 1; mk < 16; mk <<= 1) rs += __shfl_xor(rs, mk, 32);
            lrow[r] = lrow[r] * alpha[r] + rs;
            for (int u = 0; u < 4; ++u) o[u][r] *= alpha[r];
        }

        // bounce P through per-wave LDS to re-layout D-fragment -> A-fragment
        for (int u = 0; u < 4; ++u)
            for (int r = 0; r < 8; ++r)
                ps[(r + lhalf * 8) * KS_STRIDE + u * 16 + l16] = (bf16)s[u][r];
        asm volatile("s_wait_dscnt 0x0" ::: "memory");

        // O += P @ V  (M=16, N=64 dims, K=64 keys); B[k=key][n=dim] = Vt[dim][key]
        for (int kc = 0; kc < 2; ++kc) {
            v16bf pf;
            for (int e = 0; e < 16; ++e) pf[e] = ps[l16 * KS_STRIDE + kc * 32 + kmapA(e, lhalf)];
            for (int u = 0; u < 4; ++u) {
                v16bf bfr;
                const bf16* vp = vt + (u * 16 + l16) * KS_STRIDE + kc * 32 + lhalf * 16;
                for (int e = 0; e < 16; ++e) bfr[e] = vp[e];
                o[u] = __builtin_amdgcn_wmma_f32_16x16x32_bf16(
                    false, pf, false, bfr, (short)0, o[u], false, false);
            }
        }
        __syncthreads();  // all waves done with buf[i&1] before chunk i+2 lands
    }

    // normalize, apply sigmoid(gate), store bf16
    for (int u = 0; u < 4; ++u) {
        const int c = h * D_HEAD + u * 16 + l16;
        for (int r = 0; r < 8; ++r) {
            const size_t tok = tokBase + qrow0 + r + lhalf * 8;
            const float val = o[u][r] / lrow[r];
            const float g = gate[tok * D_MODEL + c];
            const float sg = 1.f / (1.f + __expf(-g));
            oh[tok * D_MODEL + c] = (bf16)(val * sg);
        }
    }
}

// ---------------------------------------------------------------------------
extern "C" void kernel_launch(void* const* d_in, const int* in_sizes, int n_in,
                              void* d_out, int out_size, void* d_ws, size_t ws_size,
                              hipStream_t stream) {
    const float* x  = (const float*)d_in[0];
    const float* Wq = (const float*)d_in[1];
    const float* Wk = (const float*)d_in[2];
    const float* Wv = (const float*)d_in[3];
    const float* Wo = (const float*)d_in[4];

    size_t off = 0;
    auto take = [&](size_t bytes) -> void* {
        void* p = (void*)((char*)d_ws + off);
        off += (bytes + 255) & ~(size_t)255;
        return p;
    };
    bf16*  xh   = (bf16*)take((size_t)NTOK * D_MODEL * 2);
    bf16*  wqT  = (bf16*)take((size_t)2 * D_MODEL * D_MODEL * 2);  // [1536][768]
    bf16*  wkT  = (bf16*)take((size_t)D_MODEL * D_MODEL * 2);      // [768][768]
    bf16*  wvT  = (bf16*)take((size_t)D_MODEL * D_MODEL * 2);
    bf16*  woT  = (bf16*)take((size_t)D_MODEL * D_MODEL * 2);
    bf16*  qh   = (bf16*)take((size_t)NTOK * D_MODEL * 2);         // [tok][d]
    bf16*  kh   = (bf16*)take((size_t)NTOK * D_MODEL * 2);         // [tok][d]
    bf16*  vTg  = (bf16*)take((size_t)NTOK * D_MODEL * 2);         // [d][tok]
    float* gbuf = (float*)take((size_t)NTOK * D_MODEL * 4);
    bf16*  ohb  = (bf16*)take((size_t)NTOK * D_MODEL * 2);

    // x: linear bf16 convert
    {
        int n = NTOK * D_MODEL;
        int blocks = (n + 255) / 256;
        if (blocks > 8192) blocks = 8192;
        cvt_f32_bf16<<<blocks, 256, 0, stream>>>(x, xh, n);
    }
    // weights: transpose + convert  ([K x N] -> [N x K])
    cvt_transpose<<<dim3((2 * D_MODEL) / 32, D_MODEL / 32), 256, 0, stream>>>(Wq, wqT, D_MODEL, 2 * D_MODEL);
    cvt_transpose<<<dim3(D_MODEL / 32, D_MODEL / 32), 256, 0, stream>>>(Wk, wkT, D_MODEL, D_MODEL);
    cvt_transpose<<<dim3(D_MODEL / 32, D_MODEL / 32), 256, 0, stream>>>(Wv, wvT, D_MODEL, D_MODEL);
    cvt_transpose<<<dim3(D_MODEL / 32, D_MODEL / 32), 256, 0, stream>>>(Wo, woT, D_MODEL, D_MODEL);

    // qg = x @ Wq   [8192 x 1536]   (split epilogue: q bf16 / gate f32)
    gemm_bf16<1><<<dim3((2 * D_MODEL) / 128, NTOK / 128), 256, 0, stream>>>(
        xh, wqT, qh, gbuf, NTOK, 2 * D_MODEL, D_MODEL);
    // k = x @ Wk  (row-major),  v = x @ Wv  (transposed store -> vT [768][8192])
    gemm_bf16<0><<<dim3(D_MODEL / 128, NTOK / 128), 256, 0, stream>>>(
        xh, wkT, kh, nullptr, NTOK, D_MODEL, D_MODEL);
    gemm_bf16<3><<<dim3(D_MODEL / 128, NTOK / 128), 256, 0, stream>>>(
        xh, wvT, vTg, nullptr, NTOK, D_MODEL, D_MODEL);

    // flash attention + gate -> ohb (bf16, [tok][d])
    attn_kernel<<<dim3(SEQ / 128, BATCH * NHEAD), 256, 0, stream>>>(qh, kh, vTg, gbuf, ohb);

    // out = o' @ Wo  (f32 store)
    gemm_bf16<2><<<dim3(D_MODEL / 128, NTOK / 128), 256, 0, stream>>>(
        ohb, woT, d_out, nullptr, NTOK, D_MODEL, D_MODEL);

    (void)in_sizes; (void)n_in; (void)out_size; (void)ws_size;
}
